// ImportanceFusion_62534723829965
// MI455X (gfx1250) — matrix operand order repaired and checked by reading, not compiled, not run on hardware
//
#include <hip/hip_runtime.h>
#include <cmath>
#include <cstdint>

#define LL   4
#define CC   256
#define C2c  128
#define HH   96
#define WW   288
#define HWp  (HH*WW)          // 27648
#define TOPK 256
#define NMSR 4
#define KTOT (CC*9)           // 2304
#define OCP  144              // 129 out-channels padded to 9 tiles of 16

typedef __attribute__((ext_vector_type(16))) _Float16 v16h;
typedef __attribute__((ext_vector_type(8)))  _Float16 v8h;
typedef __attribute__((ext_vector_type(8)))  float    v8f;

// ---------------------------------------------------------------- f32 -> f16 NHWC
__global__ void k_to_f16(const float* __restrict__ nf, _Float16* __restrict__ f16, int total) {
  int idx = blockIdx.x * blockDim.x + threadIdx.x;
  if (idx >= total) return;
  int c = idx & (CC - 1);
  int p = idx >> 8;                 // p = l*HWp + pix
  int l = p / HWp;
  int pix = p - l * HWp;
  f16[idx] = (_Float16)nf[((size_t)l * CC + c) * HWp + pix];
}

// ---------------------------------------------------------------- combined conv weights (f16, [OCP][KTOT], k = tap*256 + c)
__global__ void k_wc16(const float* __restrict__ sw, const float* __restrict__ dw,
                       _Float16* __restrict__ wc, int total) {
  int idx = blockIdx.x * blockDim.x + threadIdx.x;
  if (idx >= total) return;
  int oc = idx / KTOT;
  int k  = idx - oc * KTOT;
  int t  = k >> 8;
  int c  = k & 255;
  float w = 0.f;
  if (oc == 0)            w = sw[c * 9 + t];
  else if (oc <= C2c)     w = dw[((size_t)(oc - 1) * CC + c) * 9 + t];
  wc[idx] = (_Float16)w;
}

// ---------------------------------------------------------------- implicit-GEMM conv via WMMA
// One wave computes a 32-pixel x 16-outch tile; K-loop steps 64 (two 32-chunks
// per 3x3 tap share base pointer + bounds check). 4 independent v_wmma per trip.
__global__ void k_conv_wmma(const _Float16* __restrict__ feat16,
                            const _Float16* __restrict__ wc16,
                            const float* __restrict__ sb,
                            const float* __restrict__ db,
                            float* __restrict__ scores,
                            float* __restrict__ desc) {
  const int wave = blockIdx.x * (blockDim.x >> 5) + (threadIdx.x >> 5);
  const int lane = threadIdx.x & 31;
  const int NPG  = (LL * HWp) / 32;   // 3456 pixel groups of 32
  const int OT   = OCP / 16;          // 9 outch tiles
  const int pg   = wave / OT;
  const int ot   = wave - pg * OT;
  if (pg >= NPG) return;

  const int m   = lane & 15;
  const int grp = lane >> 4;

  // Subtile pixels: p0 = pg*32 + m, p1 = p0 + 16. W%32==0 -> whole group in one
  // image row, one layer (HWp%32==0).
  const int p0   = pg * 32 + m;
  const int li   = p0 / HWp;
  const int pix0 = p0 - li * HWp;
  const int y    = pix0 / WW;
  const int x0   = pix0 - y * WW;     // second subtile at x0 + 16

  v8f acc0 = {}, acc1 = {};
  const _Float16* wrow = wc16 + (size_t)(ot * 16 + m) * KTOT;

  auto loadA = [&](const _Float16* fp) -> v16h {
    v16h a;
    v8h lo = *reinterpret_cast<const v8h*>(fp + grp * 8);        // K = grp*8 + 0..7
    v8h hi = *reinterpret_cast<const v8h*>(fp + 16 + grp * 8);   // K = 16 + grp*8 + 0..7
    #pragma unroll
    for (int i = 0; i < 8; ++i) { a[i] = lo[i]; a[i + 8] = hi[i]; }
    return a;
  };

  for (int kb = 0; kb < KTOT; kb += 64) {
    const int tap  = kb >> 8;
    const int coff = kb & 255;
    const int dy = tap / 3 - 1, dx = tap % 3 - 1;
    const int yy  = y + dy;
    const int xxa = x0 + dx;
    const int xxb = x0 + 16 + dx;
    const bool rowok = (unsigned)yy < (unsigned)HH;

    v16h a00 = {}, a01 = {}, a10 = {}, a11 = {};
    if (rowok && (unsigned)xxa < (unsigned)WW) {
      const _Float16* fp = feat16 + ((size_t)(li * HH + yy) * WW + xxa) * CC + coff;
      a00 = loadA(fp);
      a01 = loadA(fp + 32);
    }
    if (rowok && (unsigned)xxb < (unsigned)WW) {
      const _Float16* fp = feat16 + ((size_t)(li * HH + yy) * WW + xxb) * CC + coff;
      a10 = loadA(fp);
      a11 = loadA(fp + 32);
    }
    // B-fragments: lane column N = lane%16, K = kb(+32) + grp*8 + h (h<8), 16 + grp*8 + (h-8) (h>=8)
    v16h b0 = loadA(wrow + kb);
    v16h b1 = loadA(wrow + kb + 32);
    if (kb + 256 < KTOT) __builtin_prefetch((const void*)(wrow + kb + 256), 0, 0);

    acc0 = __builtin_amdgcn_wmma_f32_16x16x32_f16(false, a00, false, b0,
                                                  (short)0, acc0, false, false);
    acc1 = __builtin_amdgcn_wmma_f32_16x16x32_f16(false, a10, false, b0,
                                                  (short)0, acc1, false, false);
    acc0 = __builtin_amdgcn_wmma_f32_16x16x32_f16(false, a01, false, b1,
                                                  (short)0, acc0, false, false);
    acc1 = __builtin_amdgcn_wmma_f32_16x16x32_f16(false, a11, false, b1,
                                                  (short)0, acc1, false, false);
  }

  // C/D: lane holds out-channel N = lane%16, VGPR v holds pixel row M = v + grp*8
  const int oc = ot * 16 + (lane & 15);
  const float sbias = sb[0];
  const float dbias = (oc >= 1 && oc <= C2c) ? db[oc - 1] : 0.f;
  #pragma unroll
  for (int v = 0; v < 8; ++v) {
    const int mrow = v + grp * 8;
    const size_t pp0 = (size_t)pg * 32 + mrow;
    const size_t pp1 = pp0 + 16;
    float v0 = acc0[v], v1 = acc1[v];
    if (oc == 0) {
      scores[pp0] = 1.f / (1.f + __expf(-(v0 + sbias)));
      scores[pp1] = 1.f / (1.f + __expf(-(v1 + sbias)));
    } else if (oc <= C2c) {
      float r0 = v0 + dbias, r1 = v1 + dbias;
      desc[pp0 * C2c + (oc - 1)] = r0 > 0.f ? r0 : 0.f;
      desc[pp1 * C2c + (oc - 1)] = r1 > 0.f ? r1 : 0.f;
    }
  }
}

// ---------------------------------------------------------------- separable 9x9 max-pool
__global__ void k_rowmax(const float* __restrict__ in, float* __restrict__ out, int total) {
  int t = blockIdx.x * blockDim.x + threadIdx.x;
  if (t >= total) return;
  int x = t % WW;
  int base = t - x;
  float mx = -3.4e38f;
  for (int d = -NMSR; d <= NMSR; ++d) {
    int xx = x + d;
    if ((unsigned)xx < (unsigned)WW) mx = fmaxf(mx, in[base + xx]);
  }
  out[t] = mx;
}
__global__ void k_colmax(const float* __restrict__ in, float* __restrict__ out, int total) {
  int t = blockIdx.x * blockDim.x + threadIdx.x;
  if (t >= total) return;
  int pix = t % HWp;
  int lb  = t - pix;
  int y = pix / WW;
  int x = pix - y * WW;
  float mx = -3.4e38f;
  for (int d = -NMSR; d <= NMSR; ++d) {
    int yy = y + d;
    if ((unsigned)yy < (unsigned)HH) mx = fmaxf(mx, in[lb + yy * WW + x]);
  }
  out[t] = mx;
}
__global__ void k_eqmask(const float* __restrict__ s, const float* __restrict__ mp,
                         float* __restrict__ m, int total) {
  int t = blockIdx.x * blockDim.x + threadIdx.x;
  if (t >= total) return;
  m[t] = (s[t] == mp[t]) ? 1.f : 0.f;
}
__global__ void k_suppress(const float* __restrict__ mpmask, const float* __restrict__ s,
                           float* __restrict__ supp, float* __restrict__ ss, int total) {
  int t = blockIdx.x * blockDim.x + threadIdx.x;
  if (t >= total) return;
  float sp = mpmask[t] > 0.f ? 1.f : 0.f;
  supp[t] = sp;
  ss[t] = sp > 0.f ? 0.f : s[t];
}
__global__ void k_updatemask(float* __restrict__ m, const float* __restrict__ ss,
                             const float* __restrict__ mpss, const float* __restrict__ supp,
                             int total) {
  int t = blockIdx.x * blockDim.x + threadIdx.x;
  if (t >= total) return;
  float nm = (ss[t] == mpss[t]) ? 1.f : 0.f;
  m[t] = ((m[t] != 0.f) || (nm != 0.f && supp[t] == 0.f)) ? 1.f : 0.f;
}
__global__ void k_applymask(const float* __restrict__ m, const float* __restrict__ s,
                            float* __restrict__ out, int total) {
  int t = blockIdx.x * blockDim.x + threadIdx.x;
  if (t >= total) return;
  out[t] = (m[t] != 0.f) ? s[t] : 0.f;
}

// ---------------------------------------------------------------- top-256 per layer (scores in LDS; 110KB fits CDNA5's 320KB/WGP)
__global__ void k_topk(const float* __restrict__ nmss, int* __restrict__ idx) {
  extern __shared__ float lds[];
  float* sv = lds;                    // HWp
  float* rv = lds + HWp;              // 1024
  int*   ri = (int*)(rv + 1024);      // 1024
  const int l = blockIdx.x;
  const int tid = threadIdx.x;
  for (int i = tid; i < HWp; i += 1024) sv[i] = nmss[(size_t)l * HWp + i];
  __syncthreads();
  for (int kk = 0; kk < TOPK; ++kk) {
    float bv = -3.0e38f; int bi = 0x7fffffff;
    for (int i = tid; i < HWp; i += 1024) {
      float v = sv[i];
      if (v > bv || (v == bv && i < bi)) { bv = v; bi = i; }
    }
    rv[tid] = bv; ri[tid] = bi;
    __syncthreads();
    for (int s = 512; s > 0; s >>= 1) {
      if (tid < s) {
        float v2 = rv[tid + s]; int i2 = ri[tid + s];
        if (v2 > rv[tid] || (v2 == rv[tid] && i2 < ri[tid])) { rv[tid] = v2; ri[tid] = i2; }
      }
      __syncthreads();
    }
    if (tid == 0) { idx[l * TOPK + kk] = ri[0]; sv[ri[0]] = -3.4e38f; }
    __syncthreads();
  }
}

// ---------------------------------------------------------------- gather + L2 normalize -> q[kk][l][c]
__global__ void k_gathernorm(const float* __restrict__ desc, const int* __restrict__ idx,
                             float* __restrict__ q) {
  __shared__ float red[C2c];
  const int l  = blockIdx.x / TOPK;
  const int kk = blockIdx.x - l * TOPK;
  const int c  = threadIdx.x;
  const int pix = idx[l * TOPK + kk];
  float v = desc[((size_t)l * HWp + pix) * C2c + c];
  red[c] = v * v;
  __syncthreads();
  for (int s = 64; s > 0; s >>= 1) { if (c < s) red[c] += red[c + s]; __syncthreads(); }
  float den = fmaxf(sqrtf(red[0]), 1e-12f);
  q[((size_t)kk * LL + l) * C2c + c] = v / den;
}

// ---------------------------------------------------------------- 4x4 self-attention per keypoint -> dcor[l][c][kk]
__global__ void k_attn(const float* __restrict__ q, float* __restrict__ dcor) {
  __shared__ float qs[LL][C2c];
  __shared__ float att[LL][LL];
  const int kk = blockIdx.x;
  const int c  = threadIdx.x;
  for (int l = 0; l < LL; ++l) qs[l][c] = q[((size_t)kk * LL + l) * C2c + c];
  __syncthreads();
  if (c < 16) {
    int a = c >> 2, b = c & 3;
    float s = 0.f;
    for (int j = 0; j < C2c; ++j) s += qs[a][j] * qs[b][j];
    att[a][b] = s * 0.08838834764831845f;     // 1/sqrt(128)
  }
  __syncthreads();
  if (c < LL) {
    float mx = att[c][0];
    for (int j = 1; j < LL; ++j) mx = fmaxf(mx, att[c][j]);
    float e[LL]; float sum = 0.f;
    for (int j = 0; j < LL; ++j) { e[j] = __expf(att[c][j] - mx); sum += e[j]; }
    for (int j = 0; j < LL; ++j) att[c][j] = e[j] / sum;
  }
  __syncthreads();
  for (int l = 0; l < LL; ++l) {
    float msg = 0.f;
    for (int mc = 0; mc < LL; ++mc) msg += att[l][mc] * qs[mc][c];
    dcor[((size_t)l * C2c + c) * TOPK + kk] = 2.f * qs[l][c] + msg;
  }
}

// ---------------------------------------------------------------- conv1d 3-tap (128->128) + ReLU
__global__ void k_proj(const float* __restrict__ dcor, const float* __restrict__ pw,
                       const float* __restrict__ pb, float* __restrict__ dproj) {
  __shared__ float xin[3][C2c];
  const int l  = blockIdx.x / TOPK;
  const int k  = blockIdx.x - l * TOPK;
  const int co = threadIdx.x;
  for (int t = 0; t < 3; ++t) {
    int kk = k + t - 1;
    xin[t][co] = ((unsigned)kk < (unsigned)TOPK) ? dcor[((size_t)l * C2c + co) * TOPK + kk] : 0.f;
  }
  __syncthreads();
  float acc = pb[co];
  for (int ci = 0; ci < C2c; ++ci) {
    const float* wp = pw + ((size_t)co * C2c + ci) * 3;
    acc += wp[0] * xin[0][ci] + wp[1] * xin[1][ci] + wp[2] * xin[2][ci];
  }
  dproj[((size_t)l * C2c + co) * TOPK + k] = acc > 0.f ? acc : 0.f;
}

// ---------------------------------------------------------------- conv1d 1x1 (128->128)
__global__ void k_projS(const float* __restrict__ dproj, const float* __restrict__ ww,
                        const float* __restrict__ bb, float* __restrict__ sc) {
  __shared__ float xin[C2c];
  const int l  = blockIdx.x / TOPK;
  const int k  = blockIdx.x - l * TOPK;
  const int co = threadIdx.x;
  xin[co] = dproj[((size_t)l * C2c + co) * TOPK + k];
  __syncthreads();
  float acc = bb[co];
  for (int ci = 0; ci < C2c; ++ci) acc += ww[(size_t)co * C2c + ci] * xin[ci];
  sc[((size_t)l * C2c + co) * TOPK + k] = acc;
}

// ---------------------------------------------------------------- stable descending rank-sort along K per (l,co)
__global__ void k_sortperm(const float* __restrict__ sc, const float* __restrict__ dproj,
                           float* __restrict__ srt) {
  __shared__ float sv[TOPK];
  const int l  = blockIdx.x / C2c;
  const int co = blockIdx.x - l * C2c;
  const int i  = threadIdx.x;
  const size_t base = ((size_t)l * C2c + co) * TOPK;
  sv[i] = sc[base + i];
  float dv = dproj[base + i];
  __syncthreads();
  float key = sv[i];
  int rank = 0;
  for (int j = 0; j < TOPK; ++j) {
    float o = sv[j];
    rank += (o > key) || (o == key && j < i);
  }
  srt[base + rank] = dv;
}

// ---------------------------------------------------------------- final conv1d (128->3), minus layer0, argmin|.| -> md[l*3+o]
__global__ void k_final_md(const float* __restrict__ srt, const float* __restrict__ fw,
                           const float* __restrict__ fb, float* __restrict__ md) {
  __shared__ float dfin[12][TOPK];
  __shared__ float rv[TOPK];
  __shared__ int   ri[TOPK];
  const int k = threadIdx.x;
  for (int l = 0; l < LL; ++l)
    for (int o = 0; o < 3; ++o) {
      float acc = fb[o];
      for (int ci = 0; ci < C2c; ++ci)
        acc += fw[o * C2c + ci] * srt[((size_t)l * C2c + ci) * TOPK + k];
      dfin[l * 3 + o][k] = acc;
    }
  __syncthreads();
  for (int o = 0; o < 3; ++o) {
    float d0 = dfin[o][k];
    for (int l = 0; l < LL; ++l) dfin[l * 3 + o][k] -= d0;
  }
  __syncthreads();
  for (int r = 0; r < 12; ++r) {
    rv[k] = fabsf(dfin[r][k]); ri[k] = k;
    __syncthreads();
    for (int s = 128; s > 0; s >>= 1) {
      if (k < s) {
        float v2 = rv[k + s]; int i2 = ri[k + s];
        if (v2 < rv[k] || (v2 == rv[k] && i2 < ri[k])) { rv[k] = v2; ri[k] = i2; }
      }
      __syncthreads();
    }
    if (k == 0) {
      float val = dfin[r][ri[0]];
      md[r] = (val == 0.f) ? 0.f : fabsf(val);   // val*(val/|val|) == |val|
    }
    __syncthreads();
  }
}

// ---------------------------------------------------------------- rotated bilinear grid sample of original feats
__global__ void k_gridsample(const float* __restrict__ nf, const float* __restrict__ md,
                             float* __restrict__ xs, long total) {
  long idx = (long)blockIdx.x * blockDim.x + threadIdx.x;
  if (idx >= total) return;
  int pix = (int)(idx % HWp);
  long t2 = idx / HWp;
  int c = (int)(t2 % CC);
  int l = (int)(t2 / CC);
  int y = pix / WW;
  int x = pix - y * WW;

  float tx = md[l * 3 + 0], ty = md[l * 3 + 1], ang = md[l * 3 + 2];
  float xsn = (2.f * x + 1.f) / WW - 1.f;
  float ysn = (2.f * y + 1.f) / HH - 1.f;
  float cs = cosf(ang), sn = sinf(ang);
  float gx = cs * xsn - sn * ysn + tx;
  float gy = sn * xsn + cs * ysn + ty;
  float ix = ((gx + 1.f) * WW - 1.f) * 0.5f;
  float iy = ((gy + 1.f) * HH - 1.f) * 0.5f;
  float x0f = floorf(ix), y0f = floorf(iy);
  float wx = ix - x0f, wy = iy - y0f;
  int x0 = (int)x0f, y0 = (int)y0f;

  const float* im = nf + ((size_t)l * CC + c) * HWp;
  auto samp = [&](int yi, int xi) -> float {
    bool valid = (xi >= 0 && xi < WW && yi >= 0 && yi < HH);
    int xc = xi < 0 ? 0 : (xi > WW - 1 ? WW - 1 : xi);
    int yc = yi < 0 ? 0 : (yi > HH - 1 ? HH - 1 : yi);
    return valid ? im[yc * WW + xc] : 0.f;
  };
  float v00 = samp(y0, x0),     v01 = samp(y0, x0 + 1);
  float v10 = samp(y0 + 1, x0), v11 = samp(y0 + 1, x0 + 1);
  xs[((size_t)l * CC + c) * HWp + pix] =
      v00 * (1.f - wx) * (1.f - wy) + v01 * wx * (1.f - wy) +
      v10 * (1.f - wx) * wy + v11 * wx * wy;
}

// ---------------------------------------------------------------- fused per-pixel sampling-attention mask + Shapley fusion
__global__ void k_fused(const float* __restrict__ xs, const float* __restrict__ mlw,
                        const float* __restrict__ mlb, const float* __restrict__ cw,
                        const float* __restrict__ cbp, float* __restrict__ out) {
  __shared__ float smlp[CC], sctr[CC];
  const int tid = threadIdx.x;
  smlp[tid] = mlw[tid];
  sctr[tid] = cw[tid];
  __syncthreads();
  const int pix = blockIdx.x * 256 + tid;

  float dk[3] = {0, 0, 0}, mk[3] = {0, 0, 0}, dl[4] = {0, 0, 0, 0};
  for (int c = 0; c < CC; ++c) {
    float f0 = xs[((size_t)0 * CC + c) * HWp + pix];
    float f1 = xs[((size_t)1 * CC + c) * HWp + pix];
    float f2 = xs[((size_t)2 * CC + c) * HWp + pix];
    float f3 = xs[((size_t)3 * CC + c) * HWp + pix];
    dk[0] += f0 * f1; dk[1] += f0 * f2; dk[2] += f0 * f3;
    float mw = smlp[c];
    mk[0] += mw * f1; mk[1] += mw * f2; mk[2] += mw * f3;
    float w2 = sctr[c];
    dl[0] += w2 * f0; dl[1] += w2 * f1; dl[2] += w2 * f2; dl[3] += w2 * f3;
  }
  const float inv16 = 1.f / 16.f;   // 1/sqrt(256)
  float s0 = dk[0] * inv16, s1 = dk[1] * inv16, s2 = dk[2] * inv16;
  float mx = fmaxf(s0, fmaxf(s1, s2));
  float e0 = __expf(s0 - mx), e1 = __expf(s1 - mx), e2 = __expf(s2 - mx);
  float inv = 1.f / (e0 + e1 + e2);
  float ctx = (e0 * mk[0] + e1 * mk[1] + e2 * mk[2]) * inv + mlb[0];
  float rl = ctx > 0.f ? ctx : 0.f;
  float score = 1.f / (1.f + __expf(-rl));
  float maskv = score > 0.5f ? 1.f : 0.f;
  dl[1] *= maskv; dl[2] *= maskv; dl[3] *= maskv;

  const float cbv = cbp[0];
  float sv[4];
  for (int id = 0; id < 4; ++id) {
    float ci = fmaxf(dl[id] + cbv, 0.f);
    float s = ci * 0.25f;
    int oth[3]; { int t = 0; for (int j = 0; j < 4; ++j) if (j != id) oth[t++] = j; }
    for (int msk = 1; msk < 8; ++msk) {
      int card = __popc(msk);
      float sum = dl[id];
      for (int j = 0; j < 3; ++j) if ((msk >> j) & 1) sum += dl[oth[j]];
      float wgt = (card == 3) ? 0.25f : (1.f / 12.f);
      s += wgt * (fmaxf(sum + cbv, 0.f) - ci);
    }
    sv[id] = s;
  }
  float mx2 = fmaxf(fmaxf(sv[0], sv[1]), fmaxf(sv[2], sv[3]));
  float ee[4]; float ss = 0.f;
  for (int j = 0; j < 4; ++j) { ee[j] = __expf(sv[j] - mx2); ss += ee[j]; }
  float fwgt[4];
  fwgt[0] = ee[0] / ss;
  for (int j = 1; j < 4; ++j) fwgt[j] = ee[j] / ss * maskv;

  for (int c = 0; c < CC; ++c) {
    float o = fwgt[0] * xs[((size_t)0 * CC + c) * HWp + pix]
            + fwgt[1] * xs[((size_t)1 * CC + c) * HWp + pix]
            + fwgt[2] * xs[((size_t)2 * CC + c) * HWp + pix]
            + fwgt[3] * xs[((size_t)3 * CC + c) * HWp + pix];
    out[(size_t)c * HWp + pix] = o;
  }
}

// ================================================================ host
extern "C" void kernel_launch(void* const* d_in, const int* in_sizes, int n_in,
                              void* d_out, int out_size, void* d_ws, size_t ws_size,
                              hipStream_t stream) {
  const float* nf  = (const float*)d_in[0];
  const float* sw  = (const float*)d_in[1];
  const float* sb  = (const float*)d_in[2];
  const float* dw  = (const float*)d_in[3];
  const float* db  = (const float*)d_in[4];
  const float* pw  = (const float*)d_in[5];
  const float* pb  = (const float*)d_in[6];
  const float* psw = (const float*)d_in[7];
  const float* psb = (const float*)d_in[8];
  const float* fw  = (const float*)d_in[9];
  const float* fb  = (const float*)d_in[10];
  const float* mlw = (const float*)d_in[11];
  const float* mlb = (const float*)d_in[12];
  const float* cw  = (const float*)d_in[13];
  const float* cb  = (const float*)d_in[14];

  char* ws = (char*)d_ws;
  size_t off = 0;
  auto carve = [&](size_t bytes) -> char* {
    char* p = ws + off;
    off += (bytes + 255) & ~(size_t)255;
    return p;
  };
  _Float16* feat16 = (_Float16*)carve((size_t)LL * HWp * CC * 2);
  _Float16* wc16   = (_Float16*)carve((size_t)OCP * KTOT * 2);
  float* scores = (float*)carve((size_t)LL * HWp * 4);
  float* desc   = (float*)carve((size_t)LL * HWp * C2c * 4);
  float* mbuf   = (float*)carve((size_t)LL * HWp * 4);
  float* b1     = (float*)carve((size_t)LL * HWp * 4);
  float* b2     = (float*)carve((size_t)LL * HWp * 4);
  float* b3     = (float*)carve((size_t)LL * HWp * 4);
  float* b4     = (float*)carve((size_t)LL * HWp * 4);
  float* nmss   = (float*)carve((size_t)LL * HWp * 4);
  int*   tkidx  = (int*)carve((size_t)LL * TOPK * 4);
  float* qb     = (float*)carve((size_t)TOPK * LL * C2c * 4);
  float* dcor   = (float*)carve((size_t)LL * C2c * TOPK * 4);
  float* dproj  = (float*)carve((size_t)LL * C2c * TOPK * 4);
  float* scb    = (float*)carve((size_t)LL * C2c * TOPK * 4);
  float* srtb   = (float*)carve((size_t)LL * C2c * TOPK * 4);
  float* mdb    = (float*)carve(64);
  float* xsb    = (float*)carve((size_t)LL * CC * HWp * 4);
  (void)ws_size; (void)in_sizes; (void)n_in; (void)out_size;

  int total = LL * HWp * CC;
  k_to_f16<<<(total + 255) / 256, 256, 0, stream>>>(nf, feat16, total);
  total = OCP * KTOT;
  k_wc16<<<(total + 255) / 256, 256, 0, stream>>>(sw, dw, wc16, total);

  {
    int waves  = (LL * HWp / 32) * (OCP / 16);   // 31104
    int blocks = waves / 4;                       // 4 waves / 128-thread block
    k_conv_wmma<<<blocks, 128, 0, stream>>>(feat16, wc16, sb, db, scores, desc);
  }

  const int tl = LL * HWp;
  const int eb = (tl + 255) / 256;
  auto mpool = [&](const float* in, float* outp) {
    k_rowmax<<<eb, 256, 0, stream>>>(in, b1, tl);
    k_colmax<<<eb, 256, 0, stream>>>(b1, outp, tl);
  };
  mpool(scores, b2);
  k_eqmask<<<eb, 256, 0, stream>>>(scores, b2, mbuf, tl);
  for (int it = 0; it < 2; ++it) {
    mpool(mbuf, b2);
    k_suppress<<<eb, 256, 0, stream>>>(b2, scores, b3, b4, tl);
    mpool(b4, b2);
    k_updatemask<<<eb, 256, 0, stream>>>(mbuf, b4, b2, b3, tl);
  }
  k_applymask<<<eb, 256, 0, stream>>>(mbuf, scores, nmss, tl);

  size_t shmem = (size_t)HWp * 4 + 1024 * 4 + 1024 * 4;
  k_topk<<<LL, 1024, shmem, stream>>>(nmss, tkidx);

  k_gathernorm<<<LL * TOPK, C2c, 0, stream>>>(desc, tkidx, qb);
  k_attn<<<TOPK, C2c, 0, stream>>>(qb, dcor);
  k_proj<<<LL * TOPK, C2c, 0, stream>>>(dcor, pw, pb, dproj);
  k_projS<<<LL * TOPK, C2c, 0, stream>>>(dproj, psw, psb, scb);
  k_sortperm<<<LL * C2c, TOPK, 0, stream>>>(scb, dproj, srtb);
  k_final_md<<<1, TOPK, 0, stream>>>(srtb, fw, fb, mdb);

  long gtot = (long)LL * CC * HWp;
  k_gridsample<<<(int)((gtot + 255) / 256), 256, 0, stream>>>(nf, mdb, xsb, gtot);

  k_fused<<<HWp / 256, 256, 0, stream>>>(xsb, mlw, mlb, cw, cb, (float*)d_out);
}
